// Hermite2d_3779571221003
// MI455X (gfx1250) — compile-verified, matrix operand order checked by reading, not run on hardware
//
#include <hip/hip_runtime.h>
#include <hip/hip_bf16.h>
#include <math.h>

typedef __attribute__((ext_vector_type(16))) _Float16 v16h;
typedef __attribute__((ext_vector_type(8)))  float    v8f;

#define CIN    128
#define COUT   128
#define NDEG   9              // DEGREE + 1
#define HW     4096           // 64*64
#define KTOT   (CIN * NDEG)   // 1152
#define CCHUNK 32             // channels per K-chunk
#define KCHUNK (CCHUNK * NDEG)// 288 (9 wmma k-steps of 32)
#define KPAD   296            // padded LDS row (f16) -> 592B, bank-friendly
#define STILE  64             // pixels per workgroup

union FragU {
    v16h h;
    struct { uint4 lo; uint4 hi; } q;
};

// Fast device transcendentals: use gfx1250 hardware tanh / exp2 when exposed.
__device__ __forceinline__ float fast_tanh(float x) {
#if __has_builtin(__builtin_amdgcn_tanhf)
    return __builtin_amdgcn_tanhf(x);
#elif __has_builtin(__builtin_amdgcn_tanh_f32)
    return __builtin_amdgcn_tanh_f32(x);
#else
    return tanhf(x);
#endif
}
__device__ __forceinline__ float fast_exp2(float x) {
#if __has_builtin(__builtin_amdgcn_exp2f)
    return __builtin_amdgcn_exp2f(x);
#else
    return exp2f(x);
#endif
}

// ---------------------------------------------------------------------------
// Prologue: weights [C_in][C_out][9] f32 -> Wt [C_out][K=c*9+d] f16 (L2-hot)
// ---------------------------------------------------------------------------
__global__ __launch_bounds__(256)
void hermite_wt_pack(const float* __restrict__ w, _Float16* __restrict__ wt) {
    int idx = blockIdx.x * 256 + threadIdx.x;     // 16384 = 128*128
    if (idx >= CIN * COUT) return;
    int c = idx >> 7;
    int o = idx & 127;
    const float* src = w  + ((size_t)c * COUT + o) * NDEG;
    _Float16*    dst = wt + (size_t)o * KTOT + c * NDEG;
#pragma unroll
    for (int d = 0; d < NDEG; ++d) dst[d] = (_Float16)src[d];
}

// ---------------------------------------------------------------------------
// Fused kernel: basis-on-the-fly + f16 WMMA GEMM, f32 accumulate.
// Grid: 2048 blocks x 256 threads. Block = 128 C_out x 64 pixels.
// Wave w -> C_out tile w (16 outs); 4 pixel sub-tiles of 16 -> 4 accumulators.
// ---------------------------------------------------------------------------
__global__ __launch_bounds__(256)
void hermite_fused(const float* __restrict__ x,
                   const _Float16* __restrict__ wt,
                   float* __restrict__ out) {
    __shared__ __align__(16) _Float16 Bl[STILE][KPAD];   // 37,888 B

    const int wg    = blockIdx.x;          // 2048
    const int b     = wg >> 6;             // image
    const int s0    = (wg & 63) << 6;      // pixel base within image
    const int tid   = threadIdx.x;
    const int lane  = tid & 31;
    const int wave  = tid >> 5;            // 0..7 -> C_out tile
    const int obase = wave << 4;
    const int half  = lane >> 4;           // lane half (K packing)
    const int l16   = lane & 15;

    // Hermite recurrence coefficients: h_d = Ac[d]*xf*h_{d-1} - Bc[d]*h_{d-2}
    const float Ac[NDEG] = {0.f, 1.41421356f, 1.0f, 0.81649658f, 0.70710678f,
                            0.63245553f, 0.57735027f, 0.53452248f, 0.5f};
    const float Bc[NDEG] = {0.f, 0.f, 0.70710678f, 0.81649658f, 0.86602540f,
                            0.89442719f, 0.91287093f, 0.92582010f, 0.93541435f};
    const float scale  = 5.12310563f;      // sqrt(2*8+1) + 1
    const float gnorm  = 0.75112554f;      // pi^-0.25
    const float nl2e_2 = -0.72134752f;     // -log2(e)/2  (exp(-t/2)=2^(t*nl2e_2))

    v8f acc0 = {}, acc1 = {}, acc2 = {}, acc3 = {};

    // A-fragment row base in packed weights (L2-resident)
    const _Float16* wrow =
        wt + (size_t)(obase + l16) * KTOT + half * 8;

    for (int ch = 0; ch < CIN / CCHUNK; ++ch) {
        __syncthreads();   // protect Bl from previous iteration's readers

        // ---- basis: 64 pixels x 32 channels, 8 (s,c) pairs per thread ----
#pragma unroll
        for (int it = 0; it < 8; ++it) {
            int p  = it * 256 + tid;
            int sl = p & 63;                       // pixel in tile
            int cl = p >> 6;                       // channel in chunk
            int c  = ch * CCHUNK + cl;
            float xv = x[((size_t)(b * CIN + c) << 12) + s0 + sl];
            float xf = fast_tanh(xv) * scale;      // v_tanh_f32 on gfx1250
            float g  = gnorm * fast_exp2(nl2e_2 * xf * xf);  // v_exp_f32
            float hm2 = g;
            float hm1 = 1.41421356f * xf * g;
            _Float16* dst = &Bl[sl][cl * NDEG];
            dst[0] = (_Float16)hm2;
            dst[1] = (_Float16)hm1;
#pragma unroll
            for (int d = 2; d < NDEG; ++d) {
                float h = Ac[d] * xf * hm1 - Bc[d] * hm2;
                dst[d] = (_Float16)h;
                hm2 = hm1; hm1 = h;
            }
        }
        __syncthreads();

        // ---- GEMM over this chunk: 9 k-steps of 32 ----
        const _Float16* wchunk = wrow + ch * KCHUNK;
#pragma unroll
        for (int t = 0; t < 9; ++t) {
            const int kb = t * 32;
            FragU a;
            a.q.lo = *(const uint4*)(wchunk + kb);        // K kb+0..7 / +8..15
            a.q.hi = *(const uint4*)(wchunk + kb + 16);   // K kb+16..23 / +24..31

            FragU b0, b1, b2, b3;
            const _Float16* bp0 = &Bl[ 0 + l16][kb + half * 8];
            const _Float16* bp1 = &Bl[16 + l16][kb + half * 8];
            const _Float16* bp2 = &Bl[32 + l16][kb + half * 8];
            const _Float16* bp3 = &Bl[48 + l16][kb + half * 8];
            b0.q.lo = *(const uint4*)(bp0);  b0.q.hi = *(const uint4*)(bp0 + 16);
            b1.q.lo = *(const uint4*)(bp1);  b1.q.hi = *(const uint4*)(bp1 + 16);
            b2.q.lo = *(const uint4*)(bp2);  b2.q.hi = *(const uint4*)(bp2 + 16);
            b3.q.lo = *(const uint4*)(bp3);  b3.q.hi = *(const uint4*)(bp3 + 16);

            acc0 = __builtin_amdgcn_wmma_f32_16x16x32_f16(false, a.h, false, b0.h,
                                                          (short)0, acc0, false, false);
            acc1 = __builtin_amdgcn_wmma_f32_16x16x32_f16(false, a.h, false, b1.h,
                                                          (short)0, acc1, false, false);
            acc2 = __builtin_amdgcn_wmma_f32_16x16x32_f16(false, a.h, false, b2.h,
                                                          (short)0, acc2, false, false);
            acc3 = __builtin_amdgcn_wmma_f32_16x16x32_f16(false, a.h, false, b3.h,
                                                          (short)0, acc3, false, false);
        }
    }

    // ---- store: lane holds N=pixel -> 64B coalesced segments per VGPR ----
    float* op = out + (((size_t)(b * COUT + obase + half * 8)) << 12) + s0 + l16;
#pragma unroll
    for (int j = 0; j < 8; ++j) {
        size_t ro = (size_t)j << 12;     // next C_out row
        op[ro +  0] = acc0[j];
        op[ro + 16] = acc1[j];
        op[ro + 32] = acc2[j];
        op[ro + 48] = acc3[j];
    }
}

// ---------------------------------------------------------------------------
extern "C" void kernel_launch(void* const* d_in, const int* in_sizes, int n_in,
                              void* d_out, int out_size, void* d_ws, size_t ws_size,
                              hipStream_t stream) {
    const float* x = (const float*)d_in[0];   // [32,128,64,64] f32
    const float* w = (const float*)d_in[1];   // [128,128,9]   f32
    float* y       = (float*)d_out;           // [32,128,64,64] f32
    _Float16* wt   = (_Float16*)d_ws;         // 1152*128 f16 = 294,912 B

    hermite_wt_pack<<<64, 256, 0, stream>>>(w, wt);
    // 32 images * (4096/64) pixel tiles = 2048 workgroups
    hermite_fused<<<2048, 256, 0, stream>>>(x, wt, y);
}
